// ReservoirLayer_67499706024344
// MI455X (gfx1250) — matrix operand order repaired and checked by reading, not compiled, not run on hardware
//
#include <hip/hip_runtime.h>

#define T_STEPS 256
#define BATCH   64
#define RS      2048
#define IS      128
#define NBLK    64
#define BCOLS   32
#define THREADS 256
#define SSTR    (RS + 8)       // padded LDS state row stride (bf16 elems)
#define WSTR    (BCOLS + 1)    // padded LDS Win stride (f32 elems)

typedef float  v2f   __attribute__((ext_vector_type(2)));
typedef float  v8f   __attribute__((ext_vector_type(8)));
typedef __bf16 v16bf __attribute__((ext_vector_type(16)));
typedef __bf16 v8bf  __attribute__((ext_vector_type(8)));

union ABf { v16bf v; v8bf h[2]; };

__device__ __forceinline__ void grid_sync(int* cnt, int* gen) {
    __threadfence();                 // release: make this thread's stores device-visible
    __syncthreads();
    if (threadIdx.x == 0) {
        int g = __hip_atomic_load(gen, __ATOMIC_RELAXED, __HIP_MEMORY_SCOPE_AGENT);
        int prev = __hip_atomic_fetch_add(cnt, 1, __ATOMIC_ACQ_REL, __HIP_MEMORY_SCOPE_AGENT);
        if (prev == NBLK - 1) {
            __hip_atomic_store(cnt, 0, __ATOMIC_RELAXED, __HIP_MEMORY_SCOPE_AGENT);
            __hip_atomic_fetch_add(gen, 1, __ATOMIC_RELEASE, __HIP_MEMORY_SCOPE_AGENT);
        } else {
            while (__hip_atomic_load(gen, __ATOMIC_ACQUIRE, __HIP_MEMORY_SCOPE_AGENT) == g) {
                __builtin_amdgcn_s_sleep(2);
            }
        }
    }
    __syncthreads();
    __threadfence();                 // acquire: invalidate stale near caches on every wave
}

// tanh via one v_exp_f32 trans op (co-executes with WMMA); clamp prevents inf/NaN.
__device__ __forceinline__ float fast_tanh(float v) {
    v = fminf(fmaxf(v, -15.0f), 15.0f);
    const float e = __expf(2.0f * v);
    return __fdividef(e - 1.0f, e + 1.0f);
}

__global__ __launch_bounds__(THREADS, 1)
void reservoir_persistent(const float* __restrict__ x,
                          const float* __restrict__ win,
                          const float* __restrict__ wr,
                          float* __restrict__ out,
                          __bf16* __restrict__ wrT,
                          __bf16* __restrict__ stA,
                          __bf16* __restrict__ stB,
                          int* cnt, int* gen)
{
    __shared__ __bf16 sState[BATCH * SSTR];   // 64*2056*2B = 263168 B
    __shared__ float  sWinT[IS * WSTR];       // 128*33*4B  =  16896 B

    const int tid  = threadIdx.x;
    const int lane = tid & 31;
    const int wave = tid >> 5;
    const int hf   = lane >> 4;     // lane half (0: lanes 0-15, 1: lanes 16-31)
    const int l15  = lane & 15;
    const int mw   = wave >> 1;     // 0..3  : 16-row M tile
    const int nw   = wave & 1;      // 0..1  : 16-col N tile
    const int m    = mw * 16 + l15;            // A-matrix row for this lane
    const int nloc = nw * 16 + l15;            // local column 0..31
    const int n0   = blockIdx.x * BCOLS;
    const int n    = n0 + nloc;                // global column

    // ---- one-time: Wr (fp32, row-major [k][n]) -> WrT (bf16, [n][k]), packed b32 stores
    for (int kb = 0; kb < RS; kb += 2 * THREADS) {
        const int k = kb + 2 * tid;
        #pragma unroll 4
        for (int nl = 0; nl < BCOLS; ++nl) {
            union { __bf16 b[2]; unsigned int u; } p;
            p.b[0] = (__bf16)wr[(size_t)k * RS + (n0 + nl)];
            p.b[1] = (__bf16)wr[(size_t)(k + 1) * RS + (n0 + nl)];
            *(unsigned int*)&wrT[(size_t)(n0 + nl) * RS + k] = p.u;
        }
    }
    // ---- one-time: Win tile -> LDS, transposed [k][nl], padded stride
    for (int idx = tid; idx < BCOLS * IS; idx += THREADS) {
        const int nl = idx >> 7;          // /128
        const int k  = idx & (IS - 1);
        sWinT[k * WSTR + nl] = win[(size_t)(n0 + nl) * IS + k];
    }
    grid_sync(cnt, gen);   // WrT built everywhere; also block-syncs sWinT

    for (int t = 0; t < T_STEPS; ++t) {
        const __bf16* sg  = (t & 1) ? stB : stA;   // t=0 reads zeroed stA
        __bf16*       sdt = (t & 1) ? stA : stB;

        // ---- stage current bf16 state (64x2048) into LDS via async global->LDS DMA.
        // Per-lane LDS byte address = low 32 bits of the generic shared pointer
        // (ISA: LDS aperture address = addr[31:0]). Tracked by ASYNCcnt.
        for (int g = tid; g < BATCH * (RS / 8); g += THREADS) {
            const int mr = g >> 8;                     // 256 x 16B chunks per row
            const int kk = (g & 255) * 8;
            const unsigned long long gaddr =
                (unsigned long long)(uintptr_t)(sg + (size_t)g * 8);
            const unsigned int laddr =
                (unsigned int)(uintptr_t)&sState[mr * SSTR + kk];
            asm volatile("global_load_async_to_lds_b128 %0, %1, off"
                         :: "v"(laddr), "v"(gaddr) : "memory");
        }
        asm volatile("s_wait_asynccnt 0x0" ::: "memory");
        __syncthreads();

        v8f accF0 = {};   // fp32 input-projection accumulators (dual chains)
        v8f accF1 = {};
        v8f accB0 = {};   // bf16 recurrence accumulators (dual chains)
        v8f accB1 = {};

        // ---- input projection: x_t (16x128 f32) @ WinT (128x16 f32), v_wmma_f32_16x16x4_f32
        const float* xrow = x + (size_t)m * (T_STEPS * IS) + (size_t)t * IS;
        #pragma unroll
        for (int k0 = 0; k0 < IS; k0 += 8) {
            {
                v2f a = *(const v2f*)(xrow + k0 + 2 * hf); // lanes<16: K=k0,k0+1; lanes>=16: +2
                v2f b;
                b.x = sWinT[(k0 + 2 * hf)     * WSTR + nloc];
                b.y = sWinT[(k0 + 2 * hf + 1) * WSTR + nloc];
                accF0 = __builtin_amdgcn_wmma_f32_16x16x4_f32(
                    false, a, false, b, (short)0, accF0, false, false);
            }
            {
                const int k1 = k0 + 4;
                v2f a = *(const v2f*)(xrow + k1 + 2 * hf);
                v2f b;
                b.x = sWinT[(k1 + 2 * hf)     * WSTR + nloc];
                b.y = sWinT[(k1 + 2 * hf + 1) * WSTR + nloc];
                accF1 = __builtin_amdgcn_wmma_f32_16x16x4_f32(
                    false, a, false, b, (short)0, accF1, false, false);
            }
        }

        // ---- recurrence: state (16x2048 bf16) @ Wr (2048x16 bf16), v_wmma_f32_16x16x32_bf16
        const __bf16* arow = &sState[m * SSTR];
        const __bf16* brow = wrT + (size_t)n * RS;
        #pragma unroll 2
        for (int k0 = 0; k0 < RS; k0 += 64) {
            {
                ABf A, B;
                // A 16x32 bf16: lanes<16 elems0-7=K k0..k0+7, elems8-15=K k0+16..k0+23 (lanes>=16: +8)
                A.h[0] = *(const v8bf*)(arow + k0 + 8 * hf);
                A.h[1] = *(const v8bf*)(arow + k0 + 16 + 8 * hf);
                // B 32x16 bf16: lanes<16 elems0-7=K k0..k0+7, elems8-15=K k0+16..k0+23 (lanes>=16: +8) (NT stream)
                B.h[0] = __builtin_nontemporal_load((const v8bf*)(brow + k0 + 8 * hf));
                B.h[1] = __builtin_nontemporal_load((const v8bf*)(brow + k0 + 16 + 8 * hf));
                accB0 = __builtin_amdgcn_wmma_f32_16x16x32_bf16(
                    false, A.v, false, B.v, (short)0, accB0, false, false);
            }
            {
                ABf A, B;
                const int k1 = k0 + 32;
                A.h[0] = *(const v8bf*)(arow + k1 + 8 * hf);
                A.h[1] = *(const v8bf*)(arow + k1 + 16 + 8 * hf);
                B.h[0] = __builtin_nontemporal_load((const v8bf*)(brow + k1 + 8 * hf));
                B.h[1] = __builtin_nontemporal_load((const v8bf*)(brow + k1 + 16 + 8 * hf));
                accB1 = __builtin_amdgcn_wmma_f32_16x16x32_bf16(
                    false, A.v, false, B.v, (short)0, accB1, false, false);
            }
        }

        // ---- epilogue: tanh, write next state (bf16) or final output (fp32)
        const int rowBase = mw * 16 + 8 * hf;   // C/D layout: VGPR r -> row rowBase + r
        if (t == T_STEPS - 1) {
            #pragma unroll
            for (int r = 0; r < 8; ++r)
                out[(size_t)(rowBase + r) * RS + n] =
                    fast_tanh((accF0[r] + accF1[r]) + (accB0[r] + accB1[r]));
        } else {
            #pragma unroll
            for (int r = 0; r < 8; ++r)
                sdt[(size_t)(rowBase + r) * RS + n] =
                    (__bf16)fast_tanh((accF0[r] + accF1[r]) + (accB0[r] + accB1[r]));
            grid_sync(cnt, gen);
        }
    }
}

extern "C" void kernel_launch(void* const* d_in, const int* in_sizes, int n_in,
                              void* d_out, int out_size, void* d_ws, size_t ws_size,
                              hipStream_t stream) {
    (void)in_sizes; (void)n_in; (void)out_size; (void)ws_size;
    const float* x   = (const float*)d_in[0];   // (64, 256*128) f32
    const float* win = (const float*)d_in[1];   // (2048, 128)   f32
    const float* wr  = (const float*)d_in[2];   // (2048, 2048)  f32
    float* out = (float*)d_out;                 // (64, 2048)    f32

    char* ws = (char*)d_ws;
    const size_t stBytes = (size_t)BATCH * RS * 2;         // 262144 B per state buffer
    __bf16* stA = (__bf16*)(ws);                           // zero-initialized (t=0 state)
    int*    cnt = (int*)(ws + stBytes);                    // barrier counter
    int*    gen = cnt + 1;                                 // barrier generation
    __bf16* stB = (__bf16*)(ws + stBytes + 64);            // ping-pong buffer
    __bf16* wrT = (__bf16*)(ws + 2 * stBytes + 64);        // 2048x2048 bf16 (8 MB)

    // zero initial state + barrier counters (graph-capturable memset node)
    hipMemsetAsync(d_ws, 0, stBytes + 64, stream);
    reservoir_persistent<<<dim3(NBLK), dim3(THREADS), 0, stream>>>(
        x, win, wr, out, wrT, stA, stB, cnt, gen);
}
